// WaveAttTransformerClassifier_82214263980300
// MI455X (gfx1250) — matrix operand order repaired and checked by reading, not compiled, not run on hardware
//
#include <hip/hip_runtime.h>
#include <hip/hip_bf16.h>
#include <math.h>

#define BB   64
#define TT   256
#define CIN  32
#define DD   256
#define SS   3
#define HH   8
#define HD   32
#define FFD  1024
#define LL   4
#define BT   (BB*TT)
#define LN_EPS 1e-5f
#define GAMMA  0.1f
#define BIGV   1e9f

typedef _Float16 h16;
typedef __attribute__((ext_vector_type(16))) _Float16 v16h;
typedef __attribute__((ext_vector_type(8)))  _Float16 h8;
typedef __attribute__((ext_vector_type(8)))  float    v8f;

__constant__ float c_dlo[8] = {-0.010597401784997278f, 0.032883011666982945f, 0.030841381835986965f,
                               -0.18703481171888114f, -0.02798376941698385f, 0.6308807679295904f,
                               0.7148465705525415f, 0.23037781330885523f};
__constant__ float c_dhi[8] = {-0.23037781330885523f, 0.7148465705525415f, -0.6308807679295904f,
                               -0.02798376941698385f, 0.18703481171888114f, 0.030841381835986965f,
                               -0.032883011666982945f, -0.010597401784997278f};

// ---- f16 fragment loaders (2 x b128 each) ----------------------------------
// A 16x32 layout: lanes 0-15 hold K{0..15}, lanes 16-31 K{16..31};
// khalf (0/16) folded into pointer -> runs at +0 and +8.
__device__ __forceinline__ v16h loadAh(const h16* __restrict__ p)
{
    h8 x0 = *(const h8*)(p);
    h8 x1 = *(const h8*)(p + 8);
    v16h r;
#pragma unroll
    for (int e = 0; e < 8; ++e) { r[e] = x0[e]; r[e + 8] = x1[e]; }
    return r;
}
// B 32x16 layout: lanes 0-15 K=0..15, lanes 16-31 K=16..31; kbase (0/16) folded
// into pointer. B stored N x K -> 16 contiguous halves.
__device__ __forceinline__ v16h loadBh(const h16* __restrict__ p)
{
    h8 x0 = *(const h8*)(p);
    h8 x1 = *(const h8*)(p + 8);
    v16h r;
#pragma unroll
    for (int e = 0; e < 8; ++e) { r[e] = x0[e]; r[e + 8] = x1[e]; }
    return r;
}

// ---------------------------------------------------------------------------
// Batched GEMM on v_wmma_f32_16x16x32_f16, f16 operands / f32 accum+output.
// One wave = 32x32 macro-tile (2 A-frags x 2 B-frags -> 4 wmma per K chunk,
// 8 global_load_b128, no conversions).
//   C[z] = alpha * A[z] x B[z]^T' (+ bias) (+ C[z] if ACCUM)
//   A: M x K (lda);  B: N x K (ldb);  C: M x N (ldc), all element strides
//   z = blockIdx.z; ob = z/nh, hh = z%nh; operand offset = ob*bs? + hh*hs?
// M, N, K multiples of 32.
// ---------------------------------------------------------------------------
template<int ACCUM>
__global__ void k_gemm_h(const h16* __restrict__ A, const h16* __restrict__ Bm,
                         const float* __restrict__ bias, float* __restrict__ C,
                         int M, int N, int K, int lda, int ldb, int ldc,
                         long long bsA, long long bsB, long long bsC,
                         int nh, long long hsA, long long hsB, long long hsC,
                         float alpha)
{
    int z  = blockIdx.z;
    int ob = z / nh, hh = z % nh;
    const h16* Ab = A + ob * bsA + hh * hsA;
    const h16* Bb = Bm + ob * bsB + hh * hsB;
    float*     Cb = C + ob * bsC + hh * hsC;

    int lane = threadIdx.x & 31;
    int wave = threadIdx.x >> 5;
    int ntn  = N >> 5;
    int tile = blockIdx.x * (blockDim.x >> 5) + wave;
    int total = (M >> 5) * ntn;
    if (tile >= total) return;               // wave-uniform: EXEC stays all-ones
    int mt = tile / ntn, nt = tile % ntn;

    int l15   = lane & 15;
    int hi    = lane >> 4;
    int khalf = hi << 4;                     // A K-base: 0 / 16
    int kbB   = hi << 4;                     // B K-base: 0 / 16
    int row0  = (mt << 5) + l15;
    int col0  = (nt << 5) + l15;

    const h16* a0p = Ab + (long long)row0 * lda + khalf;
    const h16* a1p = a0p + (long long)16 * lda;
    const h16* b0p = Bb + (long long)col0 * ldb + kbB;
    const h16* b1p = b0p + (long long)16 * ldb;

    v8f acc00 = {}, acc01 = {}, acc10 = {}, acc11 = {};
    for (int k0 = 0; k0 < K; k0 += 32) {
        v16h af0 = loadAh(a0p + k0);
        v16h af1 = loadAh(a1p + k0);
        v16h bf0 = loadBh(b0p + k0);
        v16h bf1 = loadBh(b1p + k0);
        acc00 = __builtin_amdgcn_wmma_f32_16x16x32_f16(false, af0, false, bf0, (short)0, acc00, false, false);
        acc01 = __builtin_amdgcn_wmma_f32_16x16x32_f16(false, af0, false, bf1, (short)0, acc01, false, false);
        acc10 = __builtin_amdgcn_wmma_f32_16x16x32_f16(false, af1, false, bf0, (short)0, acc10, false, false);
        acc11 = __builtin_amdgcn_wmma_f32_16x16x32_f16(false, af1, false, bf1, (short)0, acc11, false, false);
    }

    int cA = (nt << 5) + l15, cB = cA + 16;
    float bz0 = bias ? bias[cA] : 0.0f;
    float bz1 = bias ? bias[cB] : 0.0f;
#pragma unroll
    for (int r = 0; r < 8; ++r) {
        int m0 = (mt << 5) + r + (hi << 3);  // C layout: m = vgpr + 8*(lane>=16)
        int m1 = m0 + 16;
        long long i00 = (long long)m0 * ldc + cA;
        long long i01 = (long long)m0 * ldc + cB;
        long long i10 = (long long)m1 * ldc + cA;
        long long i11 = (long long)m1 * ldc + cB;
        float v00 = acc00[r] * alpha + bz0;
        float v01 = acc01[r] * alpha + bz1;
        float v10 = acc10[r] * alpha + bz0;
        float v11 = acc11[r] * alpha + bz1;
        if (ACCUM) { Cb[i00] += v00; Cb[i01] += v01; Cb[i10] += v10; Cb[i11] += v11; }
        else       { Cb[i00]  = v00; Cb[i01]  = v01; Cb[i10]  = v10; Cb[i11]  = v11; }
    }
}

// ------------------------- conversion helpers -------------------------------
__global__ void k_cvt(const float* __restrict__ src, h16* __restrict__ dst, long long n)
{
    long long i = (long long)blockIdx.x * blockDim.x + threadIdx.x;
    if (i < n) dst[i] = (h16)src[i];
}
__global__ void k_transpose_cvt(const float* __restrict__ src, h16* __restrict__ dst,
                                int R, int Ccol)
{
    long long idx = (long long)blockIdx.x * blockDim.x + threadIdx.x;
    if (idx >= (long long)R * Ccol) return;
    int r = (int)(idx / Ccol), c = (int)(idx % Ccol);
    dst[(long long)c * R + r] = (h16)src[idx];
}
// (B,T,D) f32 -> (B,H,HD,T) f16 for the P.V GEMM B-operand
__global__ void k_vT_heads(const float* __restrict__ v, h16* __restrict__ vT)
{
    long long idx = (long long)blockIdx.x * blockDim.x + threadIdx.x;  // BT*DD
    if (idx >= (long long)BT * DD) return;
    int d = (int)(idx % DD);
    int t = (int)((idx / DD) % TT);
    int b = (int)(idx / ((long long)TT * DD));
    int h = d / HD, dd = d % HD;
    vT[(((long long)b * HH + h) * HD + dd) * TT + t] = (h16)v[idx];
}

// --------------------------- SWT (db4, a-trous) -----------------------------
__global__ void k_swt(const float* __restrict__ a_in, h16* __restrict__ det_out,
                      float* __restrict__ a_out, int step)
{
    long long idx = (long long)blockIdx.x * blockDim.x + threadIdx.x; // B*T*CIN
    if (idx >= (long long)BB * TT * CIN) return;
    int c = (int)(idx % CIN);
    int t = (int)((idx / CIN) % TT);
    long long bbase = (idx / ((long long)CIN * TT)) * (long long)TT * CIN;
    float lo = 0.f, hi = 0.f;
#pragma unroll
    for (int k = 0; k < 8; ++k) {
        int ts = t - k * step; ts %= TT; if (ts < 0) ts += TT;   // jnp.roll(+k*step)
        float v = a_in[bbase + (long long)ts * CIN + c];
        lo += c_dlo[k] * v; hi += c_dhi[k] * v;
    }
    det_out[idx] = (h16)hi; a_out[idx] = lo;
}

// --------------- row softmax over length TT; f32 in, f16 out ----------------
__global__ void k_softmax(const float* __restrict__ att, h16* __restrict__ atth)
{
    long long row = blockIdx.x;
    int t = threadIdx.x;
    const float* p = att + row * TT;
    __shared__ float red[TT];
    float v = p[t];
    red[t] = v; __syncthreads();
    for (int s = 128; s > 0; s >>= 1) { if (t < s) red[t] = fmaxf(red[t], red[t + s]); __syncthreads(); }
    float mx = red[0]; __syncthreads();
    float e = expf(v - mx);
    red[t] = e; __syncthreads();
    for (int s = 128; s > 0; s >>= 1) { if (t < s) red[t] += red[t + s]; __syncthreads(); }
    atth[row * TT + t] = (h16)(e / red[0]);
}

// ----------------------- LayerNorm; f32 in, f16 out -------------------------
__global__ void k_layernorm(const float* __restrict__ x, const float* __restrict__ g,
                            const float* __restrict__ b, h16* __restrict__ y)
{
    long long row = blockIdx.x;
    int t = threadIdx.x;
    __shared__ float red[DD];
    float v = x[row * DD + t];
    red[t] = v; __syncthreads();
    for (int s = 128; s > 0; s >>= 1) { if (t < s) red[t] += red[t + s]; __syncthreads(); }
    float mean = red[0] * (1.0f / DD); __syncthreads();
    float d = v - mean;
    red[t] = d * d; __syncthreads();
    for (int s = 128; s > 0; s >>= 1) { if (t < s) red[t] += red[t + s]; __syncthreads(); }
    float var = red[0] * (1.0f / DD);
    y[row * DD + t] = (h16)(d * rsqrtf(var + LN_EPS) * g[t] + b[t]);
}

// --------------------------- squared row norms ------------------------------
__global__ void k_rownorm2(const float* __restrict__ x, float* __restrict__ out)
{
    long long row = blockIdx.x;
    int t = threadIdx.x;
    __shared__ float red[DD];
    float v = x[row * DD + t];
    red[t] = v * v; __syncthreads();
    for (int s = 128; s > 0; s >>= 1) { if (t < s) red[t] += red[t + s]; __syncthreads(); }
    if (t == 0) out[row] = red[0];
}

// ------------------- gfeat = mean over (t, s) of Vs -------------------------
__global__ void k_meanTS(const float* __restrict__ Vs, float* __restrict__ gfeat)
{
    int idx = blockIdx.x * blockDim.x + threadIdx.x;     // B*D
    if (idx >= BB * DD) return;
    int b = idx / DD, d = idx % DD;
    float s2 = 0.f;
    for (int s = 0; s < SS; ++s) {
        const float* base = Vs + (long long)s * BT * DD + (long long)b * TT * DD + d;
        for (int t = 0; t < TT; ++t) s2 += base[(long long)t * DD];
    }
    gfeat[idx] = s2 * (1.0f / (TT * SS));
}

// ------------------------------- gating -------------------------------------
__global__ void k_gate(const float* __restrict__ gfeat, const float* __restrict__ gW,
                       const float* __restrict__ gb, float* __restrict__ gate)
{
    int b = threadIdx.x;
    if (b >= BB) return;
    float lg[SS];
    for (int s = 0; s < SS; ++s) {
        float a = gb[s];
        for (int d = 0; d < DD; ++d) a += gfeat[b * DD + d] * gW[d * SS + s];
        lg[s] = a;
    }
    float mx = fmaxf(lg[0], fmaxf(lg[1], lg[2]));
    float e[SS]; float sum = 0.f;
    for (int s = 0; s < SS; ++s) { e[s] = expf(lg[s] - mx); sum += e[s]; }
    float sum2 = 0.f;
    for (int s = 0; s < SS; ++s) { e[s] /= sum; sum2 += e[s]; }
    float inv = 1.0f / fmaxf(sum2, 1e-8f);
    for (int s = 0; s < SS; ++s) gate[b * SS + s] = e[s] * inv;
}

// ---------------------- z1 = sum_s Vs[s] * gate[b,s] ------------------------
__global__ void k_combine(const float* __restrict__ Vs, const float* __restrict__ gate,
                          float* __restrict__ z1)
{
    long long idx = (long long)blockIdx.x * blockDim.x + threadIdx.x;  // BT*DD
    if (idx >= (long long)BT * DD) return;
    int b = (int)(idx / ((long long)TT * DD));
    float acc = 0.f;
    for (int s = 0; s < SS; ++s)
        acc += Vs[(long long)s * BT * DD + idx] * gate[b * SS + s];
    z1[idx] = acc;
}

// --------------------------- soft-DTW scan ----------------------------------
// dotm holds -2*x.yT per batch (B,T,T); cost = xn2[i] + zn2[j] + dotm[i,j]
__global__ void k_softdtw(const float* __restrict__ dotm, const float* __restrict__ xn2,
                          const float* __restrict__ zn2, float* __restrict__ Dout)
{
    int b = blockIdx.x;
    int i = threadIdx.x;
    __shared__ float buf[3][TT];
    buf[0][i] = BIGV; buf[1][i] = BIGV;
    __syncthreads();
    int cur = 2;
    float Dv = 0.f;
    const float inv_g = 1.0f / GAMMA;
    for (int d = 0; d < 2 * TT - 1; ++d) {
        float* rnew = buf[cur];
        float* r1 = buf[(cur + 2) % 3];   // diagonal d-1
        float* r2 = buf[(cur + 1) % 3];   // diagonal d-2
        int j = d - i;
        bool valid = (j >= 0) && (j < TT);
        float cd = valid ? (xn2[b * TT + i] + zn2[b * TT + j] +
                            dotm[((long long)b * TT + i) * TT + j]) : BIGV;
        float r1s = (i > 0) ? r1[i - 1] : BIGV;                          // R[i-1,j]
        float r1v = r1[i];                                               // R[i,j-1]
        float r2s = (i > 0) ? r2[i - 1] : ((d == 0) ? 0.0f : BIGV);      // R[i-1,j-1]
        float m = fminf(fminf(r1s, r1v), r2s);
        float ssum = expf(-(r1s - m) * inv_g) + expf(-(r1v - m) * inv_g) +
                     expf(-(r2s - m) * inv_g);
        float smin = m - GAMMA * logf(ssum);
        float r = valid ? (cd + smin) : BIGV;
        rnew[i] = r;
        __syncthreads();
        if (d == 2 * TT - 2 && i == TT - 1) Dv = r;
        cur = (cur + 1) % 3;
    }
    if (i == TT - 1) Dout[b] = Dv;
}

// -------------- h = xp * exp(-D[b]) + sinusoidal positional encoding --------
__global__ void k_scale_pe(const float* __restrict__ xp, const float* __restrict__ Dv,
                           float* __restrict__ h)
{
    long long idx = (long long)blockIdx.x * blockDim.x + threadIdx.x;  // BT*DD
    if (idx >= (long long)BT * DD) return;
    int b = (int)(idx / ((long long)TT * DD));
    int t = (int)((idx / DD) % TT);
    int d = (int)(idx % DD);
    int i = d >> 1;
    float div = expf((float)(2 * i) * (-9.210340371976184f / (float)DD)); // -ln(1e4)/d
    float ang = (float)t * div;
    float pe = (d & 1) ? cosf(ang) : sinf(ang);
    h[idx] = xp[idx] * expf(-Dv[b]) + pe;
}

// ------------------ exact GELU; f32 in, f16 out -----------------------------
__global__ void k_gelu(const float* __restrict__ x, h16* __restrict__ y, long long n)
{
    long long idx = (long long)blockIdx.x * blockDim.x + threadIdx.x;
    if (idx >= n) return;
    float v = x[idx];
    y[idx] = (h16)(0.5f * v * (1.0f + erff(v * 0.7071067811865476f)));
}

// ---- beff = bvec @ W + b2 (folds two chained biases into one) --------------
__global__ void k_vecmat(const float* __restrict__ bvec, const float* __restrict__ W,
                         const float* __restrict__ b2, float* __restrict__ out)
{
    int n = threadIdx.x;
    float a = b2[n];
    for (int d = 0; d < DD; ++d) a += bvec[d] * W[d * DD + n];
    out[n] = a;
}

// ------------------------------ classifier ----------------------------------
__global__ void k_classifier(const h16* __restrict__ hn, const float* __restrict__ clsW,
                             const float* __restrict__ clsb, float* __restrict__ out)
{
    int b = threadIdx.x;
    if (b >= BB) return;
    float a = clsb[0];
    const h16* row = hn + ((long long)b * TT + (TT - 1)) * DD;
    for (int d = 0; d < DD; ++d) a += (float)row[d] * clsW[d];
    out[b] = a;
}

// ------------------------------ host helpers --------------------------------
static inline void gemm(hipStream_t st, const h16* A, const h16* B, const float* bias,
                        float* C, int M, int N, int K, int lda, int ldb, int ldc,
                        long long bsA, long long bsB, long long bsC,
                        int nh, long long hsA, long long hsB, long long hsC,
                        int batch, float alpha, int accum)
{
    int tiles = (M / 32) * (N / 32);
    dim3 grid((tiles + 7) / 8, 1, batch);
    dim3 blk(256);
    if (accum) k_gemm_h<1><<<grid, blk, 0, st>>>(A, B, bias, C, M, N, K, lda, ldb, ldc,
                                                 bsA, bsB, bsC, nh, hsA, hsB, hsC, alpha);
    else       k_gemm_h<0><<<grid, blk, 0, st>>>(A, B, bias, C, M, N, K, lda, ldb, ldc,
                                                 bsA, bsB, bsC, nh, hsA, hsB, hsC, alpha);
}

static inline void cvt(hipStream_t st, const float* s, h16* d, long long n)
{
    k_cvt<<<dim3((unsigned)((n + 255) / 256)), 256, 0, st>>>(s, d, n);
}
static inline void tcvt(hipStream_t st, const float* s, h16* d, int R, int Ccol)
{
    long long n = (long long)R * Ccol;
    k_transpose_cvt<<<dim3((unsigned)((n + 255) / 256)), 256, 0, st>>>(s, d, R, Ccol);
}

// full MHA given f32 q/k/v projections
static inline void run_attention(hipStream_t st, const float* qf, const float* kf,
                                 const float* vf, h16* qh, h16* kh, h16* vTh,
                                 float* att, h16* atth, float* obuf)
{
    cvt(st, qf, qh, (long long)BT * DD);
    cvt(st, kf, kh, (long long)BT * DD);
    k_vT_heads<<<dim3((unsigned)(((long long)BT * DD + 255) / 256)), 256, 0, st>>>(vf, vTh);
    // scores = (q . k^T)/sqrt(hd) per (b,h)
    gemm(st, qh, kh, nullptr, att, TT, TT, HD, DD, DD, TT,
         (long long)TT * DD, (long long)TT * DD, (long long)HH * TT * TT,
         HH, HD, HD, (long long)TT * TT, BB * HH, 0.17677669529663687f, 0);
    k_softmax<<<dim3(BB * HH * TT), dim3(TT), 0, st>>>(att, atth);
    // o = att . v  (B-operand: vT (B,H,HD,T) -> contiguous K)
    gemm(st, atth, vTh, nullptr, obuf, TT, HD, TT, TT, TT, DD,
         (long long)HH * TT * TT, (long long)TT * DD, (long long)TT * DD,
         HH, (long long)TT * TT, (long long)HD * TT, HD, BB * HH, 1.0f, 0);
}

extern "C" void kernel_launch(void* const* d_in, const int* in_sizes, int n_in,
                              void* d_out, int out_size, void* d_ws, size_t ws_size,
                              hipStream_t stream)
{
    (void)in_sizes; (void)n_in; (void)out_size; (void)ws_size;
    const float* x      = (const float*)d_in[0];
    const float* ip_W   = (const float*)d_in[1];
    const float* ip_b   = (const float*)d_in[2];
    const float* wq_W   = (const float*)d_in[3];
    const float* wq_b   = (const float*)d_in[4];
    const float* wk_W   = (const float*)d_in[5];
    const float* wk_b   = (const float*)d_in[6];
    const float* wv_W   = (const float*)d_in[7];
    const float* wv_b   = (const float*)d_in[8];
    const float* gate_W = (const float*)d_in[9];
    const float* gate_b = (const float*)d_in[10];
    const float* waq_W  = (const float*)d_in[11];
    const float* waq_b  = (const float*)d_in[12];
    const float* wak_W  = (const float*)d_in[13];
    const float* wak_b  = (const float*)d_in[14];
    const float* wav_W  = (const float*)d_in[15];
    const float* wav_b  = (const float*)d_in[16];
    const float* wao_W  = (const float*)d_in[17];
    const float* wao_b  = (const float*)d_in[18];
    const float* ln1_g  = (const float*)d_in[19];
    const float* ln1_b  = (const float*)d_in[20];
    const float* ln2_g  = (const float*)d_in[21];
    const float* ln2_b  = (const float*)d_in[22];
    const float* eq_W   = (const float*)d_in[23];
    const float* eq_b   = (const float*)d_in[24];
    const float* ek_W   = (const float*)d_in[25];
    const float* ek_b   = (const float*)d_in[26];
    const float* ev_W   = (const float*)d_in[27];
    const float* ev_b   = (const float*)d_in[28];
    const float* eo_W   = (const float*)d_in[29];
    const float* eo_b   = (const float*)d_in[30];
    const float* f1_W   = (const float*)d_in[31];
    const float* f1_b   = (const float*)d_in[32];
    const float* f2_W   = (const float*)d_in[33];
    const float* f2_b   = (const float*)d_in[34];
    const float* fn_g   = (const float*)d_in[35];
    const float* fn_b   = (const float*)d_in[36];
    const float* cls_W  = (const float*)d_in[37];
    const float* cls_b  = (const float*)d_in[38];
    float* out = (float*)d_out;

    // -------- workspace layout: f32 region then f16 region --------
    float* ws = (float*)d_ws;
    size_t o = 0;
    float* xp    = ws + o; o += (size_t)BT * DD;
    float* hbuf  = ws + o; o += (size_t)BT * DD;
    float* z1    = ws + o; o += (size_t)BT * DD;
    float* qb    = ws + o; o += (size_t)BT * DD;
    float* kbuf  = ws + o; o += (size_t)BT * DD;
    float* vb    = ws + o; o += (size_t)BT * DD;
    float* obuf  = ws + o; o += (size_t)BT * DD;
    float* Vs    = ws + o; o += (size_t)SS * BT * DD;
    float* ffb   = ws + o; o += (size_t)BT * FFD;
    float* att   = ws + o; o += (size_t)BB * HH * TT * TT;   // reused as DTW dot matrix
    float* aswt0 = ws + o; o += (size_t)BT * CIN;
    float* aswt1 = ws + o; o += (size_t)BT * CIN;
    float* weff  = ws + o; o += (size_t)9 * CIN * DD;
    float* beff  = ws + o; o += (size_t)9 * DD;
    float* gfeat = ws + o; o += (size_t)BB * DD;
    float* gatev = ws + o; o += (size_t)BB * SS;
    float* xn2   = ws + o; o += (size_t)BB * TT;
    float* zn2   = ws + o; o += (size_t)BB * TT;
    float* dval  = ws + o; o += (size_t)BB;

    h16* wsh = (h16*)(ws + o);
    size_t q = 0;
    h16* xh    = wsh + q; q += (size_t)BT * CIN;
    h16* detsh = wsh + q; q += (size_t)SS * BT * CIN;
    h16* wqh   = wsh + q; q += (size_t)SS * CIN * DD;
    h16* wkh   = wsh + q; q += (size_t)SS * CIN * DD;
    h16* wvh   = wsh + q; q += (size_t)SS * CIN * DD;
    h16* waTh  = wsh + q; q += (size_t)3 * DD * DD;
    h16* ipWth = wsh + q; q += (size_t)DD * CIN;
    h16* weffTh= wsh + q; q += (size_t)9 * DD * CIN;
    h16* waoTh = wsh + q; q += (size_t)DD * DD;
    h16* eqTh  = wsh + q; q += (size_t)LL * DD * DD;
    h16* ekTh  = wsh + q; q += (size_t)LL * DD * DD;
    h16* evTh  = wsh + q; q += (size_t)LL * DD * DD;
    h16* eoTh  = wsh + q; q += (size_t)LL * DD * DD;
    h16* f1Th  = wsh + q; q += (size_t)LL * FFD * DD;
    h16* f2Th  = wsh + q; q += (size_t)LL * DD * FFD;
    h16* qh    = wsh + q; q += (size_t)BT * DD;
    h16* kh    = wsh + q; q += (size_t)BT * DD;
    h16* vTh   = wsh + q; q += (size_t)BT * DD;
    h16* oh    = wsh + q; q += (size_t)BT * DD;
    h16* hnh   = wsh + q; q += (size_t)BT * DD;
    h16* ffh   = wsh + q; q += (size_t)BT * FFD;
    h16* atth  = wsh + q; q += (size_t)BB * HH * TT * TT;
    h16* xph   = wsh + q; q += (size_t)BT * DD;
    h16* z1h   = wsh + q; q += (size_t)BT * DD;

    // -------- stage weights in f16 (transposed to N x K) --------
    tcvt(stream, ip_W, ipWth, CIN, DD);
    tcvt(stream, wao_W, waoTh, DD, DD);
    tcvt(stream, waq_W, waTh + 0 * (size_t)DD * DD, DD, DD);
    tcvt(stream, wak_W, waTh + 1 * (size_t)DD * DD, DD, DD);
    tcvt(stream, wav_W, waTh + 2 * (size_t)DD * DD, DD, DD);
    cvt(stream, wq_W, wqh, (long long)SS * CIN * DD);
    cvt(stream, wk_W, wkh, (long long)SS * CIN * DD);
    cvt(stream, wv_W, wvh, (long long)SS * CIN * DD);
    cvt(stream, x, xh, (long long)BT * CIN);
    for (int l = 0; l < LL; ++l) {
        tcvt(stream, eq_W + (size_t)l * DD * DD,  eqTh + (size_t)l * DD * DD, DD, DD);
        tcvt(stream, ek_W + (size_t)l * DD * DD,  ekTh + (size_t)l * DD * DD, DD, DD);
        tcvt(stream, ev_W + (size_t)l * DD * DD,  evTh + (size_t)l * DD * DD, DD, DD);
        tcvt(stream, eo_W + (size_t)l * DD * DD,  eoTh + (size_t)l * DD * DD, DD, DD);
        tcvt(stream, f1_W + (size_t)l * DD * FFD, f1Th + (size_t)l * FFD * DD, DD, FFD);
        tcvt(stream, f2_W + (size_t)l * FFD * DD, f2Th + (size_t)l * DD * FFD, FFD, DD);
    }

    // -------- SWT detail coefficients (coarsest-first ordering) --------
    {
        long long nswt = (long long)BB * TT * CIN;
        dim3 g((unsigned)((nswt + 255) / 256));
        k_swt<<<g, 256, 0, stream>>>(x,     detsh + (size_t)(SS - 1) * BT * CIN, aswt0, 1);
        k_swt<<<g, 256, 0, stream>>>(aswt0, detsh + (size_t)(SS - 2) * BT * CIN, aswt1, 2);
        k_swt<<<g, 256, 0, stream>>>(aswt1, detsh + (size_t)(SS - 3) * BT * CIN, aswt0, 4);
    }

    // -------- fold per-scale projection with attention in-projection --------
    // Weff[s,p] = w{qkv}_W[s] @ wa{qkv}_W ;  Beff = w{qkv}_b[s] @ wa{qkv}_W + wa{qkv}_b
    const h16*   perWh[3] = { wqh, wkh, wvh };
    const float* perB[3]  = { wq_b, wk_b, wv_b };
    const float* inW[3]   = { waq_W, wak_W, wav_W };
    const float* inB[3]   = { waq_b, wak_b, wav_b };
    for (int s = 0; s < SS; ++s)
        for (int p = 0; p < 3; ++p) {
            int id = s * 3 + p;
            gemm(stream, perWh[p] + (size_t)s * CIN * DD, waTh + (size_t)p * DD * DD, nullptr,
                 weff + (size_t)id * CIN * DD, CIN, DD, DD, DD, DD, DD,
                 0, 0, 0, 1, 0, 0, 0, 1, 1.0f, 0);
            tcvt(stream, weff + (size_t)id * CIN * DD, weffTh + (size_t)id * DD * CIN, CIN, DD);
            k_vecmat<<<1, DD, 0, stream>>>(perB[p] + (size_t)s * DD, inW[p], inB[p],
                                           beff + (size_t)id * DD);
        }

    // -------- input projection --------
    gemm(stream, xh, ipWth, ip_b, xp, BT, DD, CIN, CIN, CIN, DD,
         0, 0, 0, 1, 0, 0, 0, 1, 1.0f, 0);

    // -------- wavelet attention per scale --------
    for (int s = 0; s < SS; ++s) {
        const h16* Wd = detsh + (size_t)s * BT * CIN;
        for (int p = 0; p < 3; ++p) {
            float* dst = (p == 0) ? qb : (p == 1) ? kbuf : vb;
            int id = s * 3 + p;
            gemm(stream, Wd, weffTh + (size_t)id * DD * CIN, beff + (size_t)id * DD,
                 dst, BT, DD, CIN, CIN, CIN, DD,
                 0, 0, 0, 1, 0, 0, 0, 1, 1.0f, 0);
        }
        run_attention(stream, qb, kbuf, vb, qh, kh, vTh, att, atth, obuf);
        cvt(stream, obuf, oh, (long long)BT * DD);
        gemm(stream, oh, waoTh, wao_b, Vs + (size_t)s * BT * DD,
             BT, DD, DD, DD, DD, DD, 0, 0, 0, 1, 0, 0, 0, 1, 1.0f, 0);
    }

    // -------- gating and combination --------
    k_meanTS<<<dim3((BB * DD + 255) / 256), 256, 0, stream>>>(Vs, gfeat);
    k_gate<<<1, BB, 0, stream>>>(gfeat, gate_W, gate_b, gatev);
    k_combine<<<dim3((unsigned)(((long long)BT * DD + 255) / 256)), 256, 0, stream>>>(Vs, gatev, z1);

    // -------- soft-DTW reweight --------
    k_rownorm2<<<dim3(BB * TT), DD, 0, stream>>>(xp, xn2);
    k_rownorm2<<<dim3(BB * TT), DD, 0, stream>>>(z1, zn2);
    cvt(stream, xp, xph, (long long)BT * DD);
    cvt(stream, z1, z1h, (long long)BT * DD);
    gemm(stream, xph, z1h, nullptr, att, TT, TT, DD, DD, DD, TT,
         (long long)TT * DD, (long long)TT * DD, (long long)TT * TT,
         1, 0, 0, 0, BB, -2.0f, 0);
    k_softdtw<<<dim3(BB), TT, 0, stream>>>(att, xn2, zn2, dval);
    k_scale_pe<<<dim3((unsigned)(((long long)BT * DD + 255) / 256)), 256, 0, stream>>>(xp, dval, hbuf);

    // -------- pre-norm transformer stack --------
    for (int l = 0; l < LL; ++l) {
        k_layernorm<<<dim3(BT), DD, 0, stream>>>(hbuf, ln1_g + (size_t)l * DD, ln1_b + (size_t)l * DD, hnh);
        gemm(stream, hnh, eqTh + (size_t)l * DD * DD, eq_b + (size_t)l * DD, qb,
             BT, DD, DD, DD, DD, DD, 0, 0, 0, 1, 0, 0, 0, 1, 1.0f, 0);
        gemm(stream, hnh, ekTh + (size_t)l * DD * DD, ek_b + (size_t)l * DD, kbuf,
             BT, DD, DD, DD, DD, DD, 0, 0, 0, 1, 0, 0, 0, 1, 1.0f, 0);
        gemm(stream, hnh, evTh + (size_t)l * DD * DD, ev_b + (size_t)l * DD, vb,
             BT, DD, DD, DD, DD, DD, 0, 0, 0, 1, 0, 0, 0, 1, 1.0f, 0);
        run_attention(stream, qb, kbuf, vb, qh, kh, vTh, att, atth, obuf);
        cvt(stream, obuf, oh, (long long)BT * DD);
        // residual add via accumulate
        gemm(stream, oh, eoTh + (size_t)l * DD * DD, eo_b + (size_t)l * DD, hbuf,
             BT, DD, DD, DD, DD, DD, 0, 0, 0, 1, 0, 0, 0, 1, 1.0f, 1);

        k_layernorm<<<dim3(BT), DD, 0, stream>>>(hbuf, ln2_g + (size_t)l * DD, ln2_b + (size_t)l * DD, hnh);
        gemm(stream, hnh, f1Th + (size_t)l * FFD * DD, f1_b + (size_t)l * FFD, ffb,
             BT, FFD, DD, DD, DD, FFD, 0, 0, 0, 1, 0, 0, 0, 1, 1.0f, 0);
        k_gelu<<<dim3((unsigned)(((long long)BT * FFD + 255) / 256)), 256, 0, stream>>>(ffb, ffh, (long long)BT * FFD);
        gemm(stream, ffh, f2Th + (size_t)l * DD * FFD, f2_b + (size_t)l * DD, hbuf,
             BT, DD, FFD, FFD, FFD, DD, 0, 0, 0, 1, 0, 0, 0, 1, 1.0f, 1);
    }

    // -------- final norm + classifier --------
    k_layernorm<<<dim3(BT), DD, 0, stream>>>(hbuf, fn_g, fn_b, hnh);
    k_classifier<<<1, BB, 0, stream>>>(hnh, cls_W, cls_b, out);
}
